// InputModule_10058813407244
// MI455X (gfx1250) — matrix operand order repaired and checked by reading, not compiled, not run on hardware
//
#include <hip/hip_runtime.h>
#include <hip/hip_bf16.h>

typedef __bf16 bf16_t;
typedef __attribute__((ext_vector_type(16))) __bf16 v16bf;
typedef __attribute__((ext_vector_type(8)))  __bf16 v8bf;
typedef __attribute__((ext_vector_type(8)))  float  v8f;

#define MDIM   512
#define NLEAF  16384
#define DLEV   8
#define NNODE  4096
#define KCH    4
#define BB     64
#define SS     16
#define POOL   (1 + NLEAF + DLEV * NNODE)

__device__ __forceinline__ float sigm(float x) { return 1.0f / (1.0f + __expf(-x)); }
__device__ __forceinline__ bf16_t tobf(float x) { return (bf16_t)x; }

// Build a 16x32 A fragment from the LDS-staged tile (per-lane two 8-bf16 chunks).
__device__ __forceinline__ v16bf mk_af(const bf16_t* sAp, int k) {
    v8bf lo = *(const v8bf*)(sAp + k);        // ds_load_b128
    v8bf hi = *(const v8bf*)(sAp + k + 8);    // ds_load_b128
    v16bf f;
    #pragma unroll
    for (int i = 0; i < 8; ++i) { f[i] = lo[i]; f[8 + i] = hi[i]; }
    return f;
}

// ---------------------------------------------------------------------------
// Generic zero (u32 words)
// ---------------------------------------------------------------------------
__global__ void k_zero(float* __restrict__ p, size_t n) {
    size_t i = (size_t)blockIdx.x * blockDim.x + threadIdx.x;
    size_t stride = (size_t)gridDim.x * blockDim.x;
    for (; i < n; i += stride) p[i] = 0.0f;
}

// ---------------------------------------------------------------------------
// Weight prep: transpose+convert f32 -> bf16
// ---------------------------------------------------------------------------
// w2t[g][m] (1536 x 1024) = m<512 ? W_ioux[m][g] : W_iouh[m-512][g]
__global__ void k_w2t(const float* __restrict__ wioux, const float* __restrict__ wiouh,
                      bf16_t* __restrict__ w2t) {
    int idx = blockIdx.x * blockDim.x + threadIdx.x;   // 1536*1024
    if (idx >= 1536 * 1024) return;
    int g = idx >> 10, m = idx & 1023;
    float v = (m < 512) ? wioux[(size_t)m * 1536 + g]
                        : wiouh[(size_t)(m - 512) * 1536 + g];
    w2t[idx] = tobf(v);
}

// out[g][m] (512x512) = in[m][g]
__global__ void k_wT512(const float* __restrict__ in, bf16_t* __restrict__ out) {
    int idx = blockIdx.x * blockDim.x + threadIdx.x;   // 512*512
    if (idx >= 512 * 512) return;
    int g = idx >> 9, m = idx & 511;
    out[idx] = tobf(in[(size_t)m * 512 + g]);
}

// straight convert (GRU weights already [gate][m] row-major)
__global__ void k_conv(const float* __restrict__ in, bf16_t* __restrict__ out, int n) {
    int i = blockIdx.x * blockDim.x + threadIdx.x;
    if (i < n) out[i] = tobf(in[i]);
}

// ---------------------------------------------------------------------------
// Leaf init: h_pool[1..1+NLEAF) = embed[leaf_word_idx]
// ---------------------------------------------------------------------------
__global__ void k_leaf(const float* __restrict__ embed, const int* __restrict__ lidx,
                       float* __restrict__ h_pool) {
    int leaf = blockIdx.x;                          // NLEAF
    const float* src = embed + (size_t)lidx[leaf] * MDIM;
    float* dst = h_pool + (size_t)(1 + leaf) * MDIM;
    for (int m = threadIdx.x; m < MDIM; m += blockDim.x) dst[m] = src[m];
}

// ---------------------------------------------------------------------------
// Per-level gather + bf16 operand build
//  A2  : 4096 x 1024  = [bf16(x) | bf16(sum_k ch_h)]
//  Ach : 16384 x 512  = bf16(ch_h[n][k])
// ---------------------------------------------------------------------------
__global__ void k_level_prep(const float* __restrict__ embed,
                             const int* __restrict__ widx,
                             const int* __restrict__ cidx,
                             const float* __restrict__ h_pool,
                             bf16_t* __restrict__ A2,
                             bf16_t* __restrict__ Ach) {
    int n = blockIdx.x;                             // NNODE
    int w  = widx[n];
    int c0 = cidx[n * 4 + 0], c1 = cidx[n * 4 + 1];
    int c2 = cidx[n * 4 + 2], c3 = cidx[n * 4 + 3];
    const float* e = embed + (size_t)w * MDIM;
    for (int m = threadIdx.x; m < MDIM; m += blockDim.x) {
        float h0 = h_pool[(size_t)c0 * MDIM + m];
        float h1 = h_pool[(size_t)c1 * MDIM + m];
        float h2 = h_pool[(size_t)c2 * MDIM + m];
        float h3 = h_pool[(size_t)c3 * MDIM + m];
        A2[(size_t)n * 1024 + m]        = tobf(e[m]);
        A2[(size_t)n * 1024 + 512 + m]  = tobf(h0 + h1 + h2 + h3);
        Ach[((size_t)n * 4 + 0) * MDIM + m] = tobf(h0);
        Ach[((size_t)n * 4 + 1) * MDIM + m] = tobf(h1);
        Ach[((size_t)n * 4 + 2) * MDIM + m] = tobf(h2);
        Ach[((size_t)n * 4 + 3) * MDIM + m] = tobf(h3);
    }
}

// ---------------------------------------------------------------------------
// WMMA GEMM: C[M x N] = A[M x Kd] (row-major bf16, lda) * Bt[N x Kd]^T
//
// Block: 256 threads = 8 waves; one block owns a 16-row x 512-col tile.
// The 16 x Kd A tile is cooperatively staged in LDS once (<= 32 KB of the
// 320 KB WGP LDS); each wave computes 16x64.  The K loop is unrolled x64
// with ping-pong B-fragment register sets (no rotation copies): while one
// set feeds 4 WMMAs, the other set's global loads are in flight.
//
// A-frag (16-bit A 16x32, this code's own k order, not the ISA table's): lane l holds row l%16,
//   K [16*(l/16)..+15] (contiguous, matching B).
// B-frag (B 32x16, same k order as A): lane l holds col l%16, rows k0+16*(l/16)..+15
//   -> 16 contiguous bf16 from the transposed operand Bt.
// C/D: lane l, VGPR r -> (M = r + 8*(l/16), N = l%16).
// ---------------------------------------------------------------------------
__global__ void __launch_bounds__(256)
k_gemm_wmma(const bf16_t* __restrict__ A, int lda,
            const bf16_t* __restrict__ Bt, int ldb,
            float* __restrict__ C, int ldc,
            int NcolBlocks, int Kd) {
    __shared__ bf16_t sA[16 * 1024];                // supports Kd up to 1024

    int bm = blockIdx.x / NcolBlocks;
    int bn = blockIdx.x % NcolBlocks;
    int row0 = bm * 16;

    // cooperative stage of the 16 x Kd A tile into LDS (16B chunks)
    int nch = Kd >> 3;                              // 8 bf16 per chunk
    for (int idx = threadIdx.x; idx < 16 * nch; idx += 256) {
        int r = idx / nch, cch = idx - r * nch;
        *(v8bf*)(sA + (size_t)r * Kd + cch * 8) =
            *(const v8bf*)(A + (size_t)(row0 + r) * lda + cch * 8);
    }
    __syncthreads();

    int wave = threadIdx.x >> 5;                    // 0..7
    int lane = threadIdx.x & 31;
    int col0 = bn * 512 + wave * 64;

    const bf16_t* sAp = sA + (size_t)(lane & 15) * Kd + ((lane >> 4) << 4);

    int bhalf = (lane >> 4) * 16;
    const bf16_t* bp0 = Bt + (size_t)(col0 +  0 + (lane & 15)) * ldb + bhalf;
    const bf16_t* bp1 = Bt + (size_t)(col0 + 16 + (lane & 15)) * ldb + bhalf;
    const bf16_t* bp2 = Bt + (size_t)(col0 + 32 + (lane & 15)) * ldb + bhalf;
    const bf16_t* bp3 = Bt + (size_t)(col0 + 48 + (lane & 15)) * ldb + bhalf;

    v8f acc0 = {}, acc1 = {}, acc2 = {}, acc3 = {};

    // prime ping buffer (K block 0)
    v16bf pA0 = *(const v16bf*)(bp0);
    v16bf pA1 = *(const v16bf*)(bp1);
    v16bf pA2 = *(const v16bf*)(bp2);
    v16bf pA3 = *(const v16bf*)(bp3);

    int k0 = 0;
    for (; k0 < Kd - 64; k0 += 64) {
        // pong loads (K block k0+32) fly over ping WMMAs
        v16bf pB0 = *(const v16bf*)(bp0 + k0 + 32);
        v16bf pB1 = *(const v16bf*)(bp1 + k0 + 32);
        v16bf pB2 = *(const v16bf*)(bp2 + k0 + 32);
        v16bf pB3 = *(const v16bf*)(bp3 + k0 + 32);
        __builtin_prefetch(bp0 + k0 + 96, 0, 1);

        v16bf af0 = mk_af(sAp, k0);
        v16bf af1 = mk_af(sAp, k0 + 32);

        acc0 = __builtin_amdgcn_wmma_f32_16x16x32_bf16(false, af0, false, pA0, (short)0, acc0, false, false);
        acc1 = __builtin_amdgcn_wmma_f32_16x16x32_bf16(false, af0, false, pA1, (short)0, acc1, false, false);
        acc2 = __builtin_amdgcn_wmma_f32_16x16x32_bf16(false, af0, false, pA2, (short)0, acc2, false, false);
        acc3 = __builtin_amdgcn_wmma_f32_16x16x32_bf16(false, af0, false, pA3, (short)0, acc3, false, false);

        // ping loads (K block k0+64) fly over pong WMMAs
        pA0 = *(const v16bf*)(bp0 + k0 + 64);
        pA1 = *(const v16bf*)(bp1 + k0 + 64);
        pA2 = *(const v16bf*)(bp2 + k0 + 64);
        pA3 = *(const v16bf*)(bp3 + k0 + 64);

        acc0 = __builtin_amdgcn_wmma_f32_16x16x32_bf16(false, af1, false, pB0, (short)0, acc0, false, false);
        acc1 = __builtin_amdgcn_wmma_f32_16x16x32_bf16(false, af1, false, pB1, (short)0, acc1, false, false);
        acc2 = __builtin_amdgcn_wmma_f32_16x16x32_bf16(false, af1, false, pB2, (short)0, acc2, false, false);
        acc3 = __builtin_amdgcn_wmma_f32_16x16x32_bf16(false, af1, false, pB3, (short)0, acc3, false, false);
    }

    // tail: last 64 of K (no further global loads)
    {
        v16bf pB0 = *(const v16bf*)(bp0 + k0 + 32);
        v16bf pB1 = *(const v16bf*)(bp1 + k0 + 32);
        v16bf pB2 = *(const v16bf*)(bp2 + k0 + 32);
        v16bf pB3 = *(const v16bf*)(bp3 + k0 + 32);

        v16bf af0 = mk_af(sAp, k0);
        v16bf af1 = mk_af(sAp, k0 + 32);

        acc0 = __builtin_amdgcn_wmma_f32_16x16x32_bf16(false, af0, false, pA0, (short)0, acc0, false, false);
        acc1 = __builtin_amdgcn_wmma_f32_16x16x32_bf16(false, af0, false, pA1, (short)0, acc1, false, false);
        acc2 = __builtin_amdgcn_wmma_f32_16x16x32_bf16(false, af0, false, pA2, (short)0, acc2, false, false);
        acc3 = __builtin_amdgcn_wmma_f32_16x16x32_bf16(false, af0, false, pA3, (short)0, acc3, false, false);

        acc0 = __builtin_amdgcn_wmma_f32_16x16x32_bf16(false, af1, false, pB0, (short)0, acc0, false, false);
        acc1 = __builtin_amdgcn_wmma_f32_16x16x32_bf16(false, af1, false, pB1, (short)0, acc1, false, false);
        acc2 = __builtin_amdgcn_wmma_f32_16x16x32_bf16(false, af1, false, pB2, (short)0, acc2, false, false);
        acc3 = __builtin_amdgcn_wmma_f32_16x16x32_bf16(false, af1, false, pB3, (short)0, acc3, false, false);
    }

    int crow = row0 + 8 * (lane >> 4);
    int ccol = col0 + (lane & 15);
    #pragma unroll
    for (int r = 0; r < 8; ++r) {
        float* cp = C + (size_t)(crow + r) * ldc + ccol;
        cp[0]  = acc0[r];
        cp[16] = acc1[r];
        cp[32] = acc2[r];
        cp[48] = acc3[r];
    }
}

// ---------------------------------------------------------------------------
// Level combine: gates + cell/state update, writes pool rows for this level
// ---------------------------------------------------------------------------
__global__ void k_level_combine(const float* __restrict__ iou,
                                const float* __restrict__ fx,
                                const float* __restrict__ fh,
                                const float* __restrict__ b_ioux, const float* __restrict__ b_iouh,
                                const float* __restrict__ b_fx,   const float* __restrict__ b_fh,
                                const int* __restrict__ cidx,
                                float* __restrict__ c_pool, float* __restrict__ h_pool,
                                int dst_row0) {
    int n = blockIdx.x;                             // NNODE
    int c0 = cidx[n * 4 + 0], c1 = cidx[n * 4 + 1];
    int c2 = cidx[n * 4 + 2], c3 = cidx[n * 4 + 3];
    size_t dst = (size_t)(dst_row0 + n) * MDIM;
    for (int m = threadIdx.x; m < MDIM; m += blockDim.x) {
        float gi = sigm(iou[(size_t)n * 1536 + m]          + b_ioux[m]        + b_iouh[m]);
        float go = sigm(iou[(size_t)n * 1536 + 512 + m]    + b_ioux[512 + m]  + b_iouh[512 + m]);
        float gu = tanhf(iou[(size_t)n * 1536 + 1024 + m]  + b_ioux[1024 + m] + b_iouh[1024 + m]);
        float fxv = fx[(size_t)n * MDIM + m] + b_fx[m] + b_fh[m];
        float c = gi * gu;
        c += sigm(fxv + fh[((size_t)n * 4 + 0) * MDIM + m]) * c_pool[(size_t)c0 * MDIM + m];
        c += sigm(fxv + fh[((size_t)n * 4 + 1) * MDIM + m]) * c_pool[(size_t)c1 * MDIM + m];
        c += sigm(fxv + fh[((size_t)n * 4 + 2) * MDIM + m]) * c_pool[(size_t)c2 * MDIM + m];
        c += sigm(fxv + fh[((size_t)n * 4 + 3) * MDIM + m]) * c_pool[(size_t)c3 * MDIM + m];
        float h = go * tanhf(c);
        c_pool[dst + m] = c;
        h_pool[dst + m] = h;
    }
}

// ---------------------------------------------------------------------------
// Context gather -> bf16 GRU input operand  (B*S rows of M)
// ---------------------------------------------------------------------------
__global__ void k_gather_ctx(const float* __restrict__ h_pool,
                             const int* __restrict__ ctx,
                             bf16_t* __restrict__ xsbf) {
    int row = blockIdx.x;                           // B*S
    const float* src = h_pool + (size_t)ctx[row] * MDIM;
    bf16_t* dst = xsbf + (size_t)row * MDIM;
    for (int m = threadIdx.x; m < MDIM; m += blockDim.x) dst[m] = tobf(src[m]);
}

// ---------------------------------------------------------------------------
// GRU gate combine for one timestep
// ---------------------------------------------------------------------------
__global__ void k_gru_combine(const float* __restrict__ xg,  // (B*S) x 1536
                              const float* __restrict__ hg,  // B x 1536
                              const float* __restrict__ b_ih,
                              const float* __restrict__ b_hh,
                              float* __restrict__ hcur,      // B x 512 (f32 state)
                              bf16_t* __restrict__ hprev_bf, // B x 512 (next GEMM A)
                              float* __restrict__ out,       // B x S x 512
                              int s_xg, int s_out, int accum) {
    int idx = blockIdx.x * blockDim.x + threadIdx.x;   // B*512
    if (idx >= BB * MDIM) return;
    int b = idx >> 9, m = idx & 511;
    size_t xrow = ((size_t)b * SS + s_xg) * 1536;
    float xr = xg[xrow + m]          + b_ih[m];
    float xz = xg[xrow + 512 + m]    + b_ih[512 + m];
    float xn = xg[xrow + 1024 + m]   + b_ih[1024 + m];
    float hr = hg[(size_t)b * 1536 + m]        + b_hh[m];
    float hz = hg[(size_t)b * 1536 + 512 + m]  + b_hh[512 + m];
    float hn = hg[(size_t)b * 1536 + 1024 + m] + b_hh[1024 + m];
    float hp = hcur[idx];
    float r = sigm(xr + hr);
    float z = sigm(xz + hz);
    float nn = tanhf(xn + r * hn);
    float h = (1.0f - z) * nn + z * hp;
    hcur[idx] = h;
    hprev_bf[idx] = tobf(h);
    float* o = out + ((size_t)b * SS + s_out) * MDIM + m;
    if (accum) *o += h; else *o = h;
}

// ---------------------------------------------------------------------------
// Host launcher
// ---------------------------------------------------------------------------
static inline void launch_gemm(const bf16_t* A, int lda, const bf16_t* Bt, int ldb,
                               float* C, int ldc, int Mr, int Nc, int Kd,
                               hipStream_t stream) {
    int Mt = Mr / 16;
    int Nb = Nc / 512;                   // 512 output cols per block (8 waves x 64)
    k_gemm_wmma<<<Mt * Nb, 256, 0, stream>>>(A, lda, Bt, ldb, C, ldc, Nb, Kd);
}

extern "C" void kernel_launch(void* const* d_in, const int* in_sizes, int n_in,
                              void* d_out, int out_size, void* d_ws, size_t ws_size,
                              hipStream_t stream) {
    const float* embed    = (const float*)d_in[0];
    const float* W_ioux   = (const float*)d_in[1];
    const float* b_ioux   = (const float*)d_in[2];
    const float* W_iouh   = (const float*)d_in[3];
    const float* b_iouh   = (const float*)d_in[4];
    const float* W_fx     = (const float*)d_in[5];
    const float* b_fx     = (const float*)d_in[6];
    const float* W_fh     = (const float*)d_in[7];
    const float* b_fh     = (const float*)d_in[8];
    const float* wih_f    = (const float*)d_in[9];
    const float* whh_f    = (const float*)d_in[10];
    const float* bih_f    = (const float*)d_in[11];
    const float* bhh_f    = (const float*)d_in[12];
    const float* wih_b    = (const float*)d_in[13];
    const float* whh_b    = (const float*)d_in[14];
    const float* bih_b    = (const float*)d_in[15];
    const float* bhh_b    = (const float*)d_in[16];
    const int*   leaf_idx = (const int*)d_in[17];
    const int*   lvl_widx = (const int*)d_in[18];
    const int*   ctx_idx  = (const int*)d_in[19];
    const int*   child    = (const int*)d_in[20];
    float* out = (float*)d_out;
    (void)in_sizes; (void)n_in; (void)out_size; (void)ws_size;

    // ---- workspace layout (256B-aligned slabs) ----
    char* ws = (char*)d_ws;
    size_t off = 0;
    auto take = [&](size_t bytes) { char* p = ws + off; off = (off + bytes + 255) & ~(size_t)255; return p; };
    float*  c_pool  = (float*) take((size_t)POOL * MDIM * 4);
    float*  h_pool  = (float*) take((size_t)POOL * MDIM * 4);
    bf16_t* w2t     = (bf16_t*)take((size_t)1536 * 1024 * 2);
    bf16_t* wfxT    = (bf16_t*)take((size_t)512 * 512 * 2);
    bf16_t* wfhT    = (bf16_t*)take((size_t)512 * 512 * 2);
    bf16_t* wihf_bf = (bf16_t*)take((size_t)1536 * 512 * 2);
    bf16_t* whhf_bf = (bf16_t*)take((size_t)1536 * 512 * 2);
    bf16_t* wihb_bf = (bf16_t*)take((size_t)1536 * 512 * 2);
    bf16_t* whhb_bf = (bf16_t*)take((size_t)1536 * 512 * 2);
    bf16_t* A2      = (bf16_t*)take((size_t)NNODE * 1024 * 2);
    bf16_t* Ach     = (bf16_t*)take((size_t)NNODE * 4 * 512 * 2);
    float*  iou     = (float*) take((size_t)NNODE * 1536 * 4);
    float*  fx      = (float*) take((size_t)NNODE * 512 * 4);
    float*  fh      = (float*) take((size_t)NNODE * 4 * 512 * 4);
    bf16_t* xsbf    = (bf16_t*)take((size_t)BB * SS * 512 * 2);
    float*  xgf     = (float*) take((size_t)BB * SS * 1536 * 4);
    float*  xgb     = (float*) take((size_t)BB * SS * 1536 * 4);
    float*  hg      = (float*) take((size_t)BB * 1536 * 4);
    float*  hcur    = (float*) take((size_t)BB * 512 * 4);
    bf16_t* hprevbf = (bf16_t*)take((size_t)BB * 512 * 2);

    // ---- init pools (c_pool/h_pool contiguous) ----
    k_zero<<<4096, 256, 0, stream>>>(c_pool, (size_t)2 * POOL * MDIM);

    // ---- weight prep ----
    k_w2t<<<(1536 * 1024 + 255) / 256, 256, 0, stream>>>(W_ioux, W_iouh, w2t);
    k_wT512<<<(512 * 512 + 255) / 256, 256, 0, stream>>>(W_fx, wfxT);
    k_wT512<<<(512 * 512 + 255) / 256, 256, 0, stream>>>(W_fh, wfhT);
    k_conv<<<(1536 * 512 + 255) / 256, 256, 0, stream>>>(wih_f, wihf_bf, 1536 * 512);
    k_conv<<<(1536 * 512 + 255) / 256, 256, 0, stream>>>(whh_f, whhf_bf, 1536 * 512);
    k_conv<<<(1536 * 512 + 255) / 256, 256, 0, stream>>>(wih_b, wihb_bf, 1536 * 512);
    k_conv<<<(1536 * 512 + 255) / 256, 256, 0, stream>>>(whh_b, whhb_bf, 1536 * 512);

    // ---- leaves ----
    k_leaf<<<NLEAF, 256, 0, stream>>>(embed, leaf_idx, h_pool);

    // ---- tree levels ----
    for (int d = 0; d < DLEV; ++d) {
        const int* widx = lvl_widx + (size_t)d * NNODE;
        const int* cidx = child + (size_t)d * NNODE * KCH;
        k_level_prep<<<NNODE, 256, 0, stream>>>(embed, widx, cidx, h_pool, A2, Ach);
        // iou = [x | sum_h] @ [W_ioux; W_iouh]
        launch_gemm(A2, 1024, w2t, 1024, iou, 1536, NNODE, 1536, 1024, stream);
        // fx = x @ W_fx   (x = first 512 cols of A2)
        launch_gemm(A2, 1024, wfxT, 512, fx, 512, NNODE, 512, 512, stream);
        // fh = ch_h @ W_fh (per-child rows)
        launch_gemm(Ach, 512, wfhT, 512, fh, 512, NNODE * 4, 512, 512, stream);
        k_level_combine<<<NNODE, 256, 0, stream>>>(iou, fx, fh, b_ioux, b_iouh, b_fx, b_fh,
                                                   cidx, c_pool, h_pool,
                                                   1 + NLEAF + d * NNODE);
    }

    // ---- GRU input gates (both directions, full sequence) ----
    k_gather_ctx<<<BB * SS, 256, 0, stream>>>(h_pool, ctx_idx, xsbf);
    launch_gemm(xsbf, 512, wihf_bf, 512, xgf, 1536, BB * SS, 1536, 512, stream);
    launch_gemm(xsbf, 512, wihb_bf, 512, xgb, 1536, BB * SS, 1536, 512, stream);

    // ---- forward direction (assigns out) ----
    k_zero<<<((BB * 512 + BB * 256) + 255) / 256, 256, 0, stream>>>(hcur, (size_t)BB * 512 + BB * 256);
    for (int t = 0; t < SS; ++t) {
        launch_gemm(hprevbf, 512, whhf_bf, 512, hg, 1536, BB, 1536, 512, stream);
        k_gru_combine<<<(BB * MDIM + 255) / 256, 256, 0, stream>>>(
            xgf, hg, bih_f, bhh_f, hcur, hprevbf, out, t, t, 0);
    }

    // ---- backward direction (accumulates into out) ----
    k_zero<<<((BB * 512 + BB * 256) + 255) / 256, 256, 0, stream>>>(hcur, (size_t)BB * 512 + BB * 256);
    for (int t = 0; t < SS; ++t) {
        int s = SS - 1 - t;
        launch_gemm(hprevbf, 512, whhb_bf, 512, hg, 1536, BB, 1536, 512, stream);
        k_gru_combine<<<(BB * MDIM + 255) / 256, 256, 0, stream>>>(
            xgb, hg, bih_b, bhh_b, hcur, hprevbf, out, s, s, 1);
    }
}
